// TripletLoss_6768868458738
// MI455X (gfx1250) — matrix operand order repaired and checked by reading, not compiled, not run on hardware
//
#include <hip/hip_runtime.h>

typedef __attribute__((ext_vector_type(16))) __bf16 v16bf;
typedef __attribute__((ext_vector_type(8)))  float  v8f;

#define NPTS    4096
#define DIM     2048
#define MARGINV 0.3f
#define INF_BITS 0x7F800000u

// LDS staging for the B strip (128 rows x 32 K of bf16, hi+lo, double buffered)
#define B_ROW_PITCH 80            // 64B data + 16B pad -> conflict-free frag reads
#define B_MAT_BYTES (128 * B_ROW_PITCH)   // 10240 per matrix (hi or lo)
#define B_PHASE_BYTES (2 * B_MAT_BYTES)   // 20480 per phase (hi + lo)
#define B_LDS_BYTES (2 * B_PHASE_BYTES)   // 40960 total

union FragU {
    v16bf v;
    uint4 u4[2];
};

__device__ __forceinline__ unsigned lds_off_of(const void* p) {
    // generic -> LDS address space; low 32 bits are the wave-relative LDS byte address
    return (unsigned)(unsigned long long)(const __attribute__((address_space(3))) char*)p;
}

__device__ __forceinline__ void async_b128_to_lds(unsigned ldsoff, const __bf16* g) {
    // gfx1250 async global->LDS copy, tracked by ASYNCcnt
    asm volatile("global_load_async_to_lds_b128 %0, %1, off"
                 :: "v"(ldsoff), "v"(g) : "memory");
}

__device__ __forceinline__ void wait_async_all() {
    asm volatile("s_wait_asynccnt 0x0" ::: "memory");
}

// ---------------------------------------------------------------------------
// Kernel 1: split fp32 -> bf16 hi + bf16 residual lo; fp32 row sq-norms;
// init per-row mined ap/an (uint-encoded float; monotonic for d >= 0).
// ---------------------------------------------------------------------------
__global__ __launch_bounds__(256)
void prep_kernel(const float* __restrict__ x,
                 __bf16* __restrict__ xhi, __bf16* __restrict__ xlo,
                 float* __restrict__ sq,
                 unsigned* __restrict__ ap_bits, unsigned* __restrict__ an_bits)
{
    const int row = blockIdx.x;
    const int t   = threadIdx.x;
    const size_t base = (size_t)row * DIM;

    float ssum = 0.0f;
#pragma unroll
    for (int e = 0; e < DIM; e += 256) {
        const int idx = e + t;
        const float v  = x[base + idx];
        const __bf16 h = (__bf16)v;
        const float hf = (float)h;
        const __bf16 l = (__bf16)(v - hf);
        xhi[base + idx] = h;
        xlo[base + idx] = l;
        ssum += v * v;
    }
#pragma unroll
    for (int m = 1; m < 32; m <<= 1)
        ssum += __shfl_xor(ssum, m, 32);

    __shared__ float red[8];
    if ((t & 31) == 0) red[t >> 5] = ssum;
    __syncthreads();
    if (t == 0) {
        float s = 0.0f;
#pragma unroll
        for (int w = 0; w < 8; ++w) s += red[w];
        sq[row]      = s;
        ap_bits[row] = 0u;
        an_bits[row] = INF_BITS;
    }
}

// ---------------------------------------------------------------------------
// Kernel 2: fused Gram GEMM (bf16x3 split, fp32 WMMA accum) + batch-hard
// mining. Block = 8 waves; macro-tile 128(M) x 128(N); each wave: 16x128.
// B strip is staged in LDS via async global->LDS copies, double buffered.
// ---------------------------------------------------------------------------
__global__ __launch_bounds__(256)
void gram_mine_kernel(const __bf16* __restrict__ xhi, const __bf16* __restrict__ xlo,
                      const float* __restrict__ sq, const int* __restrict__ labels,
                      unsigned* __restrict__ ap_bits, unsigned* __restrict__ an_bits)
{
    __shared__ char bsm[B_LDS_BYTES];

    const int wave = threadIdx.x >> 5;
    const int lane = threadIdx.x & 31;
    const int half = lane >> 4;   // 0 or 1
    const int lr   = lane & 15;

    const int m0     = blockIdx.y * 128 + wave * 16;
    const int n_base = blockIdx.x * 128;

    const unsigned lds_base = lds_off_of(bsm);

    v8f acc[8];
#pragma unroll
    for (int t = 0; t < 8; ++t)
#pragma unroll
        for (int e = 0; e < 8; ++e)
            acc[t][e] = 0.0f;

    const size_t arow = (size_t)(m0 + lr) * DIM;

    // Issue one K-chunk (32 wide) of the 128-row B strip into LDS phase buffer.
    auto issue_chunk = [&](int k0, int phase) {
        const unsigned pbase = lds_base + (unsigned)(phase * B_PHASE_BYTES);
#pragma unroll
        for (int i = 0; i < 2; ++i) {
            const int c   = (int)threadIdx.x + 256 * i;  // 512 16B-chunks per matrix
            const int row = c >> 2;
            const int sub = c & 3;
            const size_t goff = (size_t)(n_base + row) * DIM + k0 + sub * 8;
            const unsigned l0 = pbase + (unsigned)(row * B_ROW_PITCH + sub * 16);
            async_b128_to_lds(l0,               xhi + goff);
            async_b128_to_lds(l0 + B_MAT_BYTES, xlo + goff);
        }
    };

    issue_chunk(0, 0);

    for (int kc = 0; kc < DIM / 32; ++kc) {
        const int k0    = kc * 32;
        const int phase = kc & 1;

        wait_async_all();     // this wave's chunk kc is in LDS
        __syncthreads();      // everyone's chunk kc is in LDS; prev phase fully read
        if (kc + 1 < DIM / 32)
            issue_chunk(k0 + 32, (kc + 1) & 1);   // stream next chunk during compute

        // A fragments (16x32 bf16) straight from global (L2-resident)
        FragU ahi, alo;
        {
            const uint4* pah = (const uint4*)(xhi + arow + k0 + half * 8);
            const uint4* pal = (const uint4*)(xlo + arow + k0 + half * 8);
            ahi.u4[0] = pah[0]; ahi.u4[1] = pah[2];
            alo.u4[0] = pal[0]; alo.u4[1] = pal[2];
        }

        const char* pb = bsm + phase * B_PHASE_BYTES;
#pragma unroll
        for (int t = 0; t < 8; ++t) {
            const int boff = (t * 16 + lr) * B_ROW_PITCH + half * 16;
            const uint4* ph = (const uint4*)(pb + boff);
            const uint4* pl = (const uint4*)(pb + boff + B_MAT_BYTES);
            FragU bhi, blo;
            bhi.u4[0] = ph[0]; bhi.u4[1] = ph[2];
            blo.u4[0] = pl[0]; blo.u4[1] = pl[2];

            acc[t] = __builtin_amdgcn_wmma_f32_16x16x32_bf16(
                false, ahi.v, false, bhi.v, (short)0, acc[t], false, false);
            acc[t] = __builtin_amdgcn_wmma_f32_16x16x32_bf16(
                false, ahi.v, false, blo.v, (short)0, acc[t], false, false);
            acc[t] = __builtin_amdgcn_wmma_f32_16x16x32_bf16(
                false, alo.v, false, bhi.v, (short)0, acc[t], false, false);
        }
    }

    // --- epilogue: distances + batch-hard mining --------------------------
    float sq_r[8];
    int   lab_r[8];
#pragma unroll
    for (int v = 0; v < 8; ++v) {
        const int r = m0 + v + half * 8;
        sq_r[v]  = sq[r];
        lab_r[v] = labels[r];
    }
    float sq_j[8];
    int   lab_j[8];
#pragma unroll
    for (int t = 0; t < 8; ++t) {
        const int j = n_base + t * 16 + lr;
        sq_j[t]  = sq[j];
        lab_j[t] = labels[j];
    }

    float ap[8], an[8];
#pragma unroll
    for (int v = 0; v < 8; ++v) { ap[v] = 0.0f; an[v] = __uint_as_float(INF_BITS); }

#pragma unroll
    for (int t = 0; t < 8; ++t) {
#pragma unroll
        for (int v = 0; v < 8; ++v) {
            const float s  = acc[t][v];
            float d2 = sq_r[v] + sq_j[t] - 2.0f * s;
            d2 = fmaxf(d2, 0.0f);
            const float d = (d2 > 0.0f) ? sqrtf(d2) : 0.0f;
            if (lab_r[v] == lab_j[t]) ap[v] = fmaxf(ap[v], d);
            else                      an[v] = fminf(an[v], d);
        }
    }

    // reduce across the 16 lanes sharing each output row (xor masks < 16 stay
    // inside the half-wave, so rows don't mix)
#pragma unroll
    for (int m = 1; m < 16; m <<= 1) {
#pragma unroll
        for (int v = 0; v < 8; ++v) {
            ap[v] = fmaxf(ap[v], __shfl_xor(ap[v], m, 32));
            an[v] = fminf(an[v], __shfl_xor(an[v], m, 32));
        }
    }

    if (lr == 0) {
#pragma unroll
        for (int v = 0; v < 8; ++v) {
            const int r = m0 + v + half * 8;
            atomicMax(&ap_bits[r], __float_as_uint(ap[v]));
            atomicMin(&an_bits[r], __float_as_uint(an[v]));
        }
    }
}

// ---------------------------------------------------------------------------
// Kernel 3: final margin-ranking reduction over anchors.
// ---------------------------------------------------------------------------
__global__ __launch_bounds__(256)
void loss_kernel(const unsigned* __restrict__ ap_bits,
                 const unsigned* __restrict__ an_bits,
                 float* __restrict__ out)
{
    const int t = threadIdx.x;
    float lsum = 0.0f, lcnt = 0.0f;
    for (int i = t; i < NPTS; i += 256) {
        const unsigned ab = ap_bits[i];
        const unsigned nb = an_bits[i];
        const float ap = __uint_as_float(ab);
        const float an = __uint_as_float(nb);
        if ((ap > 0.0f) && (nb != INF_BITS)) {
            lsum += fmaxf(ap - an + MARGINV, 0.0f);
            lcnt += 1.0f;
        }
    }
#pragma unroll
    for (int m = 1; m < 32; m <<= 1) {
        lsum += __shfl_xor(lsum, m, 32);
        lcnt += __shfl_xor(lcnt, m, 32);
    }
    __shared__ float s1[8], s2[8];
    if ((t & 31) == 0) { s1[t >> 5] = lsum; s2[t >> 5] = lcnt; }
    __syncthreads();
    if (t == 0) {
        float a = 0.0f, b = 0.0f;
#pragma unroll
        for (int w = 0; w < 8; ++w) { a += s1[w]; b += s2[w]; }
        out[0] = (b > 0.0f) ? (a / b) : 0.0f;
    }
}

// ---------------------------------------------------------------------------
extern "C" void kernel_launch(void* const* d_in, const int* in_sizes, int n_in,
                              void* d_out, int out_size, void* d_ws, size_t ws_size,
                              hipStream_t stream) {
    (void)in_sizes; (void)n_in; (void)out_size; (void)ws_size;

    const float* features = (const float*)d_in[0];
    const int*   labels   = (const int*)d_in[1];
    float*       out      = (float*)d_out;

    // workspace layout: xhi[16MB] | xlo[16MB] | sq[16KB] | ap[16KB] | an[16KB]
    __bf16*   xhi     = (__bf16*)d_ws;
    __bf16*   xlo     = xhi + (size_t)NPTS * DIM;
    float*    sq      = (float*)(xlo + (size_t)NPTS * DIM);
    unsigned* ap_bits = (unsigned*)(sq + NPTS);
    unsigned* an_bits = ap_bits + NPTS;

    prep_kernel<<<NPTS, 256, 0, stream>>>(features, xhi, xlo, sq, ap_bits, an_bits);

    dim3 grid(NPTS / 128, NPTS / 128);
    gram_mine_kernel<<<grid, 256, 0, stream>>>(xhi, xlo, sq, labels, ap_bits, an_bits);

    loss_kernel<<<1, 256, 0, stream>>>(ap_bits, an_bits, out);
}
